// BipartiteGraphRecommender_19250043420815
// MI455X (gfx1250) — compile-verified, matrix operand order checked by reading, not run on hardware
//
#include <hip/hip_runtime.h>
#include <cstddef>

// ---------------- WMMA types / helpers (CDNA5, wave32) ----------------
typedef __attribute__((ext_vector_type(16))) __bf16 v16bf;
typedef __attribute__((ext_vector_type(8)))  __bf16 v8bf;
typedef __attribute__((ext_vector_type(8)))  float  v8f;
typedef unsigned short bf16raw;

#define WMMA_BF16(a, b, c) \
  __builtin_amdgcn_wmma_f32_16x16x32_bf16(false, (a), false, (b), (short)0, (c), false, false)

// ---- CDNA5 async global->LDS 16B copy (ASYNCcnt-tracked, bypasses VGPRs) ----
// ISA 15.18.3: GLOBAL_LOAD_ASYNC_TO_LDS_B128, GV mode:
//   vdst = wave-relative LDS byte offset, v[addr:addr+1] = 64-bit global address.
__device__ __forceinline__ void async_cp16(const void* gsrc, void* ldst) {
  typedef __attribute__((address_space(3))) char lds_char;
  unsigned lofs = (unsigned)(unsigned long long)(lds_char*)ldst;
  unsigned long long ga = (unsigned long long)gsrc;
  asm volatile("global_load_async_to_lds_b128 %0, %1, off"
               :
               : "v"(lofs), "v"(ga)
               : "memory");
}

__device__ __forceinline__ void wait_async0() {
  asm volatile("s_wait_asynccnt 0x0" ::: "memory");
}

// 16x32 fragment from row-major LDS tile (leading dim ld, elements).
// Lanes 0-15: row m0+lane, K = {k0..k0+7, k0+16..k0+23}; lanes 16-31: K = {+8..+15, +24..+31}.
// Used for A tiles (m0 = row tile) AND for B tiles staged TRANSPOSED (m0 = col tile),
// so every fragment load is two contiguous 16-byte LDS reads (ds_load_b128).
__device__ __forceinline__ v16bf frag16(const __bf16* base, int ld, int m0, int k0, int lane) {
  const int hf = (lane >> 4) & 1, m = lane & 15;
  const __bf16* p = base + (size_t)(m0 + m) * ld + k0 + hf * 8;
  v8bf lo = *reinterpret_cast<const v8bf*>(p);
  v8bf hi = *reinterpret_cast<const v8bf*>(p + 16);
  return __builtin_shufflevector(lo, hi, 0, 1, 2, 3, 4, 5, 6, 7, 8, 9, 10, 11, 12, 13, 14, 15);
}

// C layout: VGPR j -> row = r0 + half*8 + j, col = c0 + lane%16

// ---------------- K0: id histogram (mask + segment-mean degrees) ----------------
__global__ void k_hist(const int* __restrict__ u, const int* __restrict__ it,
                       int* __restrict__ cu, int* __restrict__ ci, int B_) {
  int t = blockIdx.x * blockDim.x + threadIdx.x;
  if (t < B_) {
    atomicAdd(&cu[u[t]], 1);
    atomicAdd(&ci[it[t]], 1);
  }
}

// ---------------- K1: projection GEMM  x = relu(maskedEmb @ W + b) ----------------
// A: [128 x 64] (ld 72), W^T: [128 cols x 64 k] (ld 72)
__global__ __launch_bounds__(256) void k_proj(
    const float* __restrict__ embU, const float* __restrict__ embI,
    const int* __restrict__ cu, const int* __restrict__ ci,
    const float* __restrict__ W, const float* __restrict__ bias,
    bf16raw* __restrict__ Xr, int NU_, int Ntot) {
  __bf16* X = reinterpret_cast<__bf16*>(Xr);
  __shared__ alignas(16) __bf16 sA[128 * 72];
  __shared__ alignas(16) __bf16 sWT[128 * 72];
  __shared__ float sBias[128];
  const int tid = threadIdx.x;
  const int r0 = blockIdx.x * 128;
  if (tid < 128) sBias[tid] = bias[tid];
  for (int i = tid; i < 64 * 128; i += 256) {   // W is [64 k x 128 c]; coalesced read, transpose on write
    int k = i >> 7, c = i & 127;
    sWT[c * 72 + k] = (__bf16)W[i];
  }
  for (int i = tid; i < 128 * 64; i += 256) {
    int rl = i >> 6, k = i & 63, r = r0 + rl;
    float v = 0.f;
    if (r < Ntot) {
      if (r < NU_) { if (cu[r] > 0) v = embU[(size_t)r * 64 + k]; }
      else { int ii = r - NU_; if (ci[ii] > 0) v = embI[(size_t)ii * 64 + k]; }
    }
    sA[rl * 72 + k] = (__bf16)v;
  }
  __syncthreads();
  const int wave = tid >> 5, lane = tid & 31, rw = wave * 16;
  v8f acc[8];
#pragma unroll
  for (int t = 0; t < 8; ++t)
#pragma unroll
    for (int j = 0; j < 8; ++j) acc[t][j] = 0.f;
#pragma unroll
  for (int kb = 0; kb < 2; ++kb) {
    v16bf a = frag16(sA, 72, rw, kb * 32, lane);
#pragma unroll
    for (int ct = 0; ct < 8; ++ct) {
      v16bf b = frag16(sWT, 72, ct * 16, kb * 32, lane);
      acc[ct] = WMMA_BF16(a, b, acc[ct]);
    }
  }
  const int hf = (lane >> 4) & 1, n = lane & 15;
#pragma unroll
  for (int ct = 0; ct < 8; ++ct)
#pragma unroll
    for (int j = 0; j < 8; ++j) {
      int row = r0 + rw + hf * 8 + j, col = ct * 16 + n;
      if (row < Ntot) {
        float v = acc[ct][j] + sBias[col];
        v = v > 0.f ? v : 0.f;
        X[(size_t)row * 128 + col] = (__bf16)v;
      }
    }
}

// ---------------- K2: edge scatter-sum (one wave per edge, both directions) ----------------
__global__ __launch_bounds__(256) void k_agg(
    const bf16raw* __restrict__ Xr, float* __restrict__ AGG,
    const int* __restrict__ u, const int* __restrict__ it, int NU_, int B_) {
  const __bf16* X = reinterpret_cast<const __bf16*>(Xr);
  int g = blockIdx.x * blockDim.x + threadIdx.x;
  int e = g >> 5, lane = g & 31;
  if (e >= B_) return;
  int un = u[e];
  int in_ = NU_ + it[e];
  int c0 = lane * 4;
#pragma unroll
  for (int q = 0; q < 4; ++q) {
    float xu = (float)X[(size_t)un * 128 + c0 + q];
    float xi = (float)X[(size_t)in_ * 128 + c0 + q];
    atomicAdd(&AGG[(size_t)in_ * 128 + c0 + q], xu);
    atomicAdd(&AGG[(size_t)un * 128 + c0 + q], xi);
  }
}

// ---------------- K3: fused SAGE layer: x = relu(bn((agg/deg)@wl + bl + x@wr)), in-place ----------------
// A1/A2: [128 x 32] (ld 40); W^T slices: [128 cols x 32 k] (ld 40)
__global__ __launch_bounds__(256) void k_sage(
    const float* __restrict__ AGG, bf16raw* __restrict__ Xr,
    const int* __restrict__ cu, const int* __restrict__ ci,
    const float* __restrict__ wl, const float* __restrict__ bl,
    const float* __restrict__ wr, const float* __restrict__ g,
    const float* __restrict__ bb, int NU_, int Ntot) {
  __bf16* X = reinterpret_cast<__bf16*>(Xr);
  __shared__ alignas(16) __bf16 sA1[128 * 40], sA2[128 * 40];
  __shared__ alignas(16) __bf16 sW1T[128 * 40], sW2T[128 * 40];
  __shared__ float sInv[128], sScale[128], sBeta[128], sBl[128];
  const int tid = threadIdx.x, r0 = blockIdx.x * 128;
  const float bn_inv = 0.99999500003749968f;  // 1/sqrt(1 + 1e-5)
  if (tid < 128) {
    int r = r0 + tid, deg = 1;
    if (r < Ntot) deg = (r < NU_) ? cu[r] : ci[r - NU_];
    sInv[tid] = 1.0f / (float)(deg > 1 ? deg : 1);
    sScale[tid] = g[tid] * bn_inv;
    sBeta[tid] = bb[tid];
    sBl[tid] = bl[tid];
  }
  const int wave = tid >> 5, lane = tid & 31, rw = wave * 16;
  v8f acc[8];
#pragma unroll
  for (int t = 0; t < 8; ++t)
#pragma unroll
    for (int j = 0; j < 8; ++j) acc[t][j] = 0.f;

  for (int kb = 0; kb < 4; ++kb) {
    const int k0 = kb * 32;
    __syncthreads();  // sInv ready (iter 0) / previous frag reads done (iter >0)
    // x tile: untransformed bf16 copy -> async DMA to LDS (512 x 16B chunks / kb)
    for (int q = tid; q < 512; q += 256) {
      int rl = q >> 2, part = q & 3, r = r0 + rl;
      __bf16* ldst = sA2 + rl * 40 + part * 8;
      if (r < Ntot) {
        async_cp16(X + (size_t)r * 128 + k0 + part * 8, ldst);
      } else {
#pragma unroll
        for (int t = 0; t < 8; ++t) ldst[t] = (__bf16)0.f;
      }
    }
    for (int i = tid; i < 128 * 32; i += 256) {
      int rl = i >> 5, k = i & 31, r = r0 + rl;
      float a1 = (r < Ntot) ? AGG[(size_t)r * 128 + k0 + k] * sInv[rl] : 0.f;
      sA1[rl * 40 + k] = (__bf16)a1;
    }
    for (int i = tid; i < 32 * 128; i += 256) {  // coalesced read of [32 k x 128 c] slice, transpose on write
      int k = i >> 7, c = i & 127;
      sW1T[c * 40 + k] = (__bf16)wl[(size_t)(k0 + k) * 128 + c];
      sW2T[c * 40 + k] = (__bf16)wr[(size_t)(k0 + k) * 128 + c];
    }
    wait_async0();
    __syncthreads();
    v16bf a1 = frag16(sA1, 40, rw, 0, lane);
    v16bf a2 = frag16(sA2, 40, rw, 0, lane);
#pragma unroll
    for (int ct = 0; ct < 8; ++ct) {
      v16bf b1 = frag16(sW1T, 40, ct * 16, 0, lane);
      acc[ct] = WMMA_BF16(a1, b1, acc[ct]);
      v16bf b2 = frag16(sW2T, 40, ct * 16, 0, lane);
      acc[ct] = WMMA_BF16(a2, b2, acc[ct]);
    }
  }
  const int hf = (lane >> 4) & 1, n = lane & 15;
#pragma unroll
  for (int ct = 0; ct < 8; ++ct)
#pragma unroll
    for (int j = 0; j < 8; ++j) {
      int row = r0 + rw + hf * 8 + j, col = ct * 16 + n;
      if (row < Ntot) {
        float v = acc[ct][j] + sBl[col];
        v = sScale[col] * v + sBeta[col];
        v = v > 0.f ? v : 0.f;
        X[(size_t)row * 128 + col] = (__bf16)v;
      }
    }
}

// ---------------- K4: output GEMM  xo = x @ out_w + out_b  (fp32 out) ----------------
// A: [128 x 128] (ld 136); W^T: [64 cols x 128 k] (ld 136)
__global__ __launch_bounds__(256) void k_out(
    const bf16raw* __restrict__ Xr, const float* __restrict__ W,
    const float* __restrict__ bias, float* __restrict__ XO, int Ntot) {
  const __bf16* X = reinterpret_cast<const __bf16*>(Xr);
  __shared__ alignas(16) __bf16 sA[128 * 136];
  __shared__ alignas(16) __bf16 sWT[64 * 136];
  __shared__ float sBias[64];
  const int tid = threadIdx.x, r0 = blockIdx.x * 128;
  if (tid < 64) sBias[tid] = bias[tid];
  for (int i = tid; i < 128 * 64; i += 256) {  // W is [128 k x 64 c]; coalesced read, transpose on write
    int k = i >> 6, c = i & 63;
    sWT[c * 136 + k] = (__bf16)W[i];
  }
  // x tile: 128 rows x 16 chunks of 16B -> async DMA to LDS
  for (int q = tid; q < 128 * 16; q += 256) {
    int rl = q >> 4, part = q & 15, r = r0 + rl;
    __bf16* ldst = sA + rl * 136 + part * 8;
    if (r < Ntot) {
      async_cp16(X + (size_t)r * 128 + part * 8, ldst);
    } else {
#pragma unroll
      for (int t = 0; t < 8; ++t) ldst[t] = (__bf16)0.f;
    }
  }
  wait_async0();
  __syncthreads();
  const int wave = tid >> 5, lane = tid & 31, rw = wave * 16;
  v8f acc[4];
#pragma unroll
  for (int t = 0; t < 4; ++t)
#pragma unroll
    for (int j = 0; j < 8; ++j) acc[t][j] = 0.f;
#pragma unroll
  for (int kb = 0; kb < 4; ++kb) {
    v16bf a = frag16(sA, 136, rw, kb * 32, lane);
#pragma unroll
    for (int ct = 0; ct < 4; ++ct) {
      v16bf b = frag16(sWT, 136, ct * 16, kb * 32, lane);
      acc[ct] = WMMA_BF16(a, b, acc[ct]);
    }
  }
  const int hf = (lane >> 4) & 1, n = lane & 15;
#pragma unroll
  for (int ct = 0; ct < 4; ++ct)
#pragma unroll
    for (int j = 0; j < 8; ++j) {
      int row = r0 + rw + hf * 8 + j, col = ct * 16 + n;
      if (row < Ntot) XO[(size_t)row * 64 + col] = acc[ct][j] + sBias[col];
    }
}

// ---------------- K5: predictor: gather u/it, h=relu([u,it]@W1+b1), pred=sigmoid(h.w2+b2) ----------------
// A: [128 x 128] (ld 136); W1^T slices: [128 cols x 32 k] (ld 40)
__global__ __launch_bounds__(256) void k_pred(
    const float* __restrict__ XO, const int* __restrict__ uids,
    const int* __restrict__ iids, const float* __restrict__ W1,
    const float* __restrict__ b1, const float* __restrict__ W2,
    const float* __restrict__ b2, float* __restrict__ pred,
    float* __restrict__ uo, float* __restrict__ io, int NU_, int B_) {
  __shared__ alignas(16) __bf16 sA[128 * 136];
  __shared__ alignas(16) __bf16 sWT[128 * 40];
  __shared__ int   sU[128], sI[128];
  __shared__ float sB1[128], sW2[128], sPred[128];
  const int tid = threadIdx.x, r0 = blockIdx.x * 128;
  if (tid < 128) {
    int p = r0 + tid;
    sU[tid] = (p < B_) ? uids[p] : 0;
    sI[tid] = (p < B_) ? iids[p] : 0;
    sB1[tid] = b1[tid];
    sW2[tid] = W2[tid];
    sPred[tid] = 0.f;
  }
  __syncthreads();
  for (int i = tid; i < 128 * 128; i += 256) {
    int rl = i >> 7, c = i & 127, p = r0 + rl;
    float v = 0.f;
    if (p < B_) {
      if (c < 64) {
        v = XO[(size_t)sU[rl] * 64 + c];
        uo[(size_t)p * 64 + c] = v;
      } else {
        v = XO[(size_t)(NU_ + sI[rl]) * 64 + (c - 64)];
        io[(size_t)p * 64 + (c - 64)] = v;
      }
    }
    sA[rl * 136 + c] = (__bf16)v;
  }
  const int wave = tid >> 5, lane = tid & 31, rw = wave * 16;
  v8f acc[8];
#pragma unroll
  for (int t = 0; t < 8; ++t)
#pragma unroll
    for (int j = 0; j < 8; ++j) acc[t][j] = 0.f;
  for (int kb = 0; kb < 4; ++kb) {
    __syncthreads();  // covers sA staging (kb==0) and prior sWT frag reads (kb>0)
    for (int i = tid; i < 32 * 128; i += 256) {  // coalesced read of [32 k x 128 c] slice of W1
      int k = i >> 7, c = i & 127;
      sWT[c * 40 + k] = (__bf16)W1[(size_t)(kb * 32 + k) * 128 + c];
    }
    __syncthreads();
    v16bf a = frag16(sA, 136, rw, kb * 32, lane);
#pragma unroll
    for (int ct = 0; ct < 8; ++ct) {
      v16bf b = frag16(sWT, 40, ct * 16, 0, lane);
      acc[ct] = WMMA_BF16(a, b, acc[ct]);
    }
  }
  __syncthreads();
  const int hf = (lane >> 4) & 1, n = lane & 15;
#pragma unroll
  for (int ct = 0; ct < 8; ++ct)
#pragma unroll
    for (int j = 0; j < 8; ++j) {
      int rowl = rw + hf * 8 + j, col = ct * 16 + n;
      float hv = acc[ct][j] + sB1[col];
      hv = hv > 0.f ? hv : 0.f;
      atomicAdd(&sPred[rowl], hv * sW2[col]);  // ds_add_f32 reduction for h . w2
    }
  __syncthreads();
  if (tid < 128) {
    int p = r0 + tid;
    if (p < B_) {
      float s = sPred[tid] + b2[0];
      pred[p] = 1.f / (1.f + __expf(-s));
    }
  }
}

// ---------------- host launch ----------------
extern "C" void kernel_launch(void* const* d_in, const int* in_sizes, int n_in,
                              void* d_out, int out_size, void* d_ws, size_t ws_size,
                              hipStream_t stream) {
  (void)n_in; (void)out_size; (void)ws_size;
  const float* embU  = (const float*)d_in[0];
  const float* embI  = (const float*)d_in[1];
  const float* projw = (const float*)d_in[2];
  const float* projb = (const float*)d_in[3];
  const float* s1wl  = (const float*)d_in[4];
  const float* s1bl  = (const float*)d_in[5];
  const float* s1wr  = (const float*)d_in[6];
  const float* bn1g  = (const float*)d_in[7];
  const float* bn1b  = (const float*)d_in[8];
  const float* s2wl  = (const float*)d_in[9];
  const float* s2bl  = (const float*)d_in[10];
  const float* s2wr  = (const float*)d_in[11];
  const float* bn2g  = (const float*)d_in[12];
  const float* bn2b  = (const float*)d_in[13];
  const float* outw  = (const float*)d_in[14];
  const float* outb  = (const float*)d_in[15];
  const float* pw1   = (const float*)d_in[16];
  const float* pb1   = (const float*)d_in[17];
  const float* pw2   = (const float*)d_in[18];
  const float* pb2   = (const float*)d_in[19];
  const int*   uids  = (const int*)d_in[20];
  const int*   iids  = (const int*)d_in[21];

  const int NUv  = in_sizes[0] / 64;
  const int NIv  = in_sizes[1] / 64;
  const int Bv   = in_sizes[20];
  const int Ntot = NUv + NIv;

  // workspace layout: AGG fp32 [N*128] (reused as XO [N*64]) | X bf16 [N*128] | counts
  char* ws = (char*)d_ws;
  float* AGG = (float*)ws;
  size_t aggBytes = (size_t)Ntot * 128 * sizeof(float);
  bf16raw* X = (bf16raw*)(ws + aggBytes);
  size_t xBytes = (size_t)Ntot * 128 * sizeof(bf16raw);
  int* cu = (int*)(ws + aggBytes + xBytes);
  int* ci = cu + NUv;

  const int nodeBlocks = (Ntot + 127) / 128;

  (void)hipMemsetAsync(cu, 0, (size_t)(NUv + NIv) * sizeof(int), stream);
  k_hist<<<(Bv + 255) / 256, 256, 0, stream>>>(uids, iids, cu, ci, Bv);

  k_proj<<<nodeBlocks, 256, 0, stream>>>(embU, embI, cu, ci, projw, projb, X, NUv, Ntot);

  // SAGE layer 1
  (void)hipMemsetAsync(AGG, 0, aggBytes, stream);
  k_agg<<<(Bv + 7) / 8, 256, 0, stream>>>(X, AGG, uids, iids, NUv, Bv);
  k_sage<<<nodeBlocks, 256, 0, stream>>>(AGG, X, cu, ci, s1wl, s1bl, s1wr, bn1g, bn1b, NUv, Ntot);

  // SAGE layer 2
  (void)hipMemsetAsync(AGG, 0, aggBytes, stream);
  k_agg<<<(Bv + 7) / 8, 256, 0, stream>>>(X, AGG, uids, iids, NUv, Bv);
  k_sage<<<nodeBlocks, 256, 0, stream>>>(AGG, X, cu, ci, s2wl, s2bl, s2wr, bn2g, bn2b, NUv, Ntot);

  // output projection -> XO (reuses AGG buffer, free after last aggregation)
  float* XO = AGG;
  k_out<<<nodeBlocks, 256, 0, stream>>>(X, outw, outb, XO, Ntot);

  // predictor + output writes
  float* pred = (float*)d_out;
  float* uo = pred + Bv;
  float* io = uo + (size_t)Bv * 64;
  k_pred<<<(Bv + 127) / 128, 256, 0, stream>>>(XO, uids, iids, pw1, pb1, pw2, pb2,
                                               pred, uo, io, NUv, Bv);
}
